// MultiHeadedAttention_LoHi_75823352643720
// MI455X (gfx1250) — matrix operand order repaired and checked by reading, not compiled, not run on hardware
//
#include <hip/hip_runtime.h>
#include <cmath>

// ---------------------------------------------------------------------------
// Types for CDNA5 WMMA (wave32)
// ---------------------------------------------------------------------------
typedef __bf16 bf16;
typedef __attribute__((ext_vector_type(8)))  bf16  v8bf;
typedef __attribute__((ext_vector_type(16))) bf16  v16bf;
typedef __attribute__((ext_vector_type(8)))  float v8f;

#define BM 128   // block tile M
#define BN 64    // block tile N
#define BK 32    // K step (one wmma K)
#define LDT 40   // padded LDS row stride (elements) to spread banks

// ---------------------------------------------------------------------------
// CDNA5 async global->LDS copy (ASYNCcnt-tracked, bypasses VGPRs).
// vdst VGPR = per-lane LDS byte offset (low 32 bits of generic LDS address),
// vaddr = 64-bit global address.  No clang builtin -> inline asm.
// ---------------------------------------------------------------------------
__device__ __forceinline__ void async_b128(unsigned lds_off, const void* g) {
  asm volatile("global_load_async_to_lds_b128 %0, %1, off"
               :: "v"(lds_off), "v"(g) : "memory");
}
__device__ __forceinline__ void wait_async0() {
  asm volatile("s_wait_asynccnt 0" ::: "memory");
}

// ---------------------------------------------------------------------------
// Fragment helpers
// 16-bit A 16x32 layout: lane<16 -> row=lane, K = {kb..kb+7, 16+kb..16+kb+7},
// kb = (lane>=16)?8:0.  B 32x16: col=lane&15, same K set as A.
// Both are two contiguous 16B chunks in a [row][K]
// K-contiguous LDS tile -> 2x ds b128 loads per fragment.
// ---------------------------------------------------------------------------
__device__ __forceinline__ v16bf frag16(const bf16* p) {
  v8bf lo = *(const v8bf*)p;
  v8bf hi = *(const v8bf*)(p + 16);
  return __builtin_shufflevector(lo, hi, 0,1,2,3,4,5,6,7,8,9,10,11,12,13,14,15);
}

__device__ __forceinline__ v8f wmma_bf16(v16bf a, v16bf b, v8f c) {
  // (neg_a, A, neg_b, B, c_mod, C, reuse_a, reuse_b)
  return __builtin_amdgcn_wmma_f32_16x16x32_bf16(false, a, false, b, (short)0, c,
                                                 false, false);
}

__device__ __forceinline__ void zero_acc(v8f acc[2][2]) {
#pragma unroll
  for (int i = 0; i < 2; ++i)
#pragma unroll
    for (int j = 0; j < 2; ++j)
#pragma unroll
      for (int r = 0; r < 8; ++r) acc[i][j][r] = 0.f;
}

// ---------------------------------------------------------------------------
// Core 128x64 GEMM:  D[m][n] = sum_k A[m][k] * B[n][k]
// A: [M][K] row-major (lda), B: [N][K] row-major (ldb), both bf16, K%32==0.
// Double-buffered LDS fed by async global->LDS copies; one barrier per K-step.
// 8 waves: wave (wm,wn) owns a 32x32 output tile (2x2 WMMA).
// ---------------------------------------------------------------------------
__device__ __forceinline__ void gemm_core(const bf16* __restrict__ A, int lda,
                                          const bf16* __restrict__ B, int ldb,
                                          int K, v8f acc[2][2]) {
  __shared__ __align__(16) bf16 ldsA[2][BM * LDT];
  __shared__ __align__(16) bf16 ldsB[2][BN * LDT];
  const int tid  = threadIdx.x;
  const int lane = tid & 31;
  const int wid  = tid >> 5;
  const int wm   = wid & 3, wn = wid >> 2;
  const int m0   = blockIdx.y * BM;
  const int n0   = blockIdx.x * BN;

  // staging coords: A tile = 128x32 (2x 16B/thread), B tile = 64x32 (1x 16B)
  const int ar = tid >> 2, as = (tid & 3) * 8;
  const int br = tid >> 2, bs = (tid & 3) * 8;

  // fragment coords
  const int frow = lane & 15;
  const int fkbA = (lane >> 4) << 3;   // {0,8}
  const int fkbB = (lane >> 4) << 3;   // {0,8}

  const unsigned aBase = (unsigned)(uintptr_t)&ldsA[0][0];
  const unsigned bBase = (unsigned)(uintptr_t)&ldsB[0][0];
  const unsigned aOff0 = (unsigned)((ar * LDT + as) * sizeof(bf16));
  const unsigned aOff1 = (unsigned)(((ar + 64) * LDT + as) * sizeof(bf16));
  const unsigned bOff  = (unsigned)((br * LDT + bs) * sizeof(bf16));
  const bf16* aSrc0 = A + (size_t)(m0 + ar) * lda + as;
  const bf16* aSrc1 = A + (size_t)(m0 + ar + 64) * lda + as;
  const bf16* bSrc  = B + (size_t)(n0 + br) * ldb + bs;

  auto issue = [&](int bufSel, int k0) {
    const unsigned da = aBase + (unsigned)bufSel * (unsigned)(BM * LDT * sizeof(bf16));
    const unsigned db = bBase + (unsigned)bufSel * (unsigned)(BN * LDT * sizeof(bf16));
    async_b128(da + aOff0, aSrc0 + k0);
    async_b128(da + aOff1, aSrc1 + k0);
    async_b128(db + bOff,  bSrc  + k0);
  };

  issue(0, 0);
  int buf = 0;
  for (int k0 = 0; k0 < K; k0 += BK, buf ^= 1) {
    wait_async0();       // this wave's copies (into current buf) have landed
    __syncthreads();     // all waves' copies visible; prior reads of buf^1 done
    if (k0 + BK < K) issue(buf ^ 1, k0 + BK);       // overlap with compute
    if (k0 + 2 * BK < K) {                          // L2 prefetch 2 steps ahead
      __builtin_prefetch(aSrc0 + k0 + 2 * BK, 0, 3);
      __builtin_prefetch(bSrc  + k0 + 2 * BK, 0, 3);
    }

    const bf16* tA = &ldsA[buf][0];
    const bf16* tB = &ldsB[buf][0];
    v16bf a0 = frag16(&tA[(wm * 32 +      frow) * LDT + fkbA]);
    v16bf a1 = frag16(&tA[(wm * 32 + 16 + frow) * LDT + fkbA]);
    v16bf b0 = frag16(&tB[(wn * 32 +      frow) * LDT + fkbB]);
    v16bf b1 = frag16(&tB[(wn * 32 + 16 + frow) * LDT + fkbB]);

    acc[0][0] = wmma_bf16(a0, b0, acc[0][0]);
    acc[0][1] = wmma_bf16(a0, b1, acc[0][1]);
    acc[1][0] = wmma_bf16(a1, b0, acc[1][0]);
    acc[1][1] = wmma_bf16(a1, b1, acc[1][1]);
  }
}

// C/D layout: vgpr r of lane L holds D[(L>=16?8:0)+r][L&15] of each 16x16 tile.
template <typename F>
__device__ __forceinline__ void epilogue_apply(const v8f (&acc)[2][2], F f) {
  const int lane = threadIdx.x & 31;
  const int wid  = threadIdx.x >> 5;
  const int wm   = wid & 3, wn = wid >> 2;
  const int coff = lane & 15;
  const int roff = (lane >> 4) << 3;
  const int mB   = blockIdx.y * BM + wm * 32;
  const int nB   = blockIdx.x * BN + wn * 32;
#pragma unroll
  for (int i = 0; i < 2; ++i)
#pragma unroll
    for (int j = 0; j < 2; ++j)
#pragma unroll
      for (int r = 0; r < 8; ++r)
        f(mB + i * 16 + roff + r, nB + j * 16 + coff, acc[i][j][r]);
}

// ---------------------------------------------------------------------------
// Scatter helpers (fused to_patches / from_patches)
// ---------------------------------------------------------------------------
__device__ __forceinline__ void scatter_qkv_lo(int m, int n, float v,
                                               const float* bias, bf16* o0,
                                               bf16* o1, int which) {
  v += bias[n];
  const int bt = m >> 10, hw = m & 1023;
  const int b = bt >> 2, t = bt & 3;
  const int cfg = n >> 9, cl = n & 511;
  const int y = hw >> 5, x = hw & 31;
  const int ls = cfg ? 1 : 2;                 // log2 patch edge (2x2 / 4x4)
  const int oy = y >> ls, py = y & ((1 << ls) - 1);
  const int ox = x >> ls, px = x & ((1 << ls) - 1);
  const int ow = 32 >> ls;                    // 8 / 16
  const int tok = (t * ow + oy) * ow + ox;    // lo tokens first
  const int dd  = (cl << (2 * ls)) + (py << ls) + px;
  const int T = 8 * ow * ow;                  // 512 / 2048
  const int D = 512 << (2 * ls);              // 8192 / 2048
  bf16* o = cfg ? o1 : o0;
  if (which < 2) o[(size_t)(b * T + tok) * D + dd] = (bf16)v;   // Q,K: [b][tok][d]
  else           o[(size_t)(b * D + dd) * T + tok] = (bf16)v;   // V:   [b][d][tok]
}

__device__ __forceinline__ void scatter_qkv_hi(int m, int n, float v,
                                               const float* bias, bf16* o0,
                                               bf16* o1, int which) {
  v += bias[n];
  const int bt = m >> 12, hw = m & 4095;
  const int b = bt >> 2, t = bt & 3;
  const int cfg = n >> 7, cl = n & 127;
  const int y = hw >> 6, x = hw & 63;
  const int ls = cfg ? 2 : 3;                 // hi patch edge 4 / 8
  const int oy = y >> ls, py = y & ((1 << ls) - 1);
  const int ox = x >> ls, px = x & ((1 << ls) - 1);
  const int ow = 64 >> ls;                    // 8 / 16 (same grid as lo)
  const int n_lo = 4 * ow * ow;
  const int tok = n_lo + (t * ow + oy) * ow + ox;
  const int dd  = (cl << (2 * ls)) + (py << ls) + px;
  const int T = 2 * n_lo;
  const int D = 128 << (2 * ls);              // 8192 / 2048
  bf16* o = cfg ? o1 : o0;
  if (which < 2) o[(size_t)(b * T + tok) * D + dd] = (bf16)v;
  else           o[(size_t)(b * D + dd) * T + tok] = (bf16)v;
}

__device__ __forceinline__ void scatter_pv(int b, int tok, int dd, float v,
                                           bf16* ylo, bf16* yhi, int cfg) {
  const int ls = cfg ? 1 : 2;
  const int ow = 32 >> ls;                    // 8 / 16
  const int n_lo = 4 * ow * ow;
  const bf16 bv = (bf16)v;
  if (tok < n_lo) {
    const int sh = 2 * (5 - ls);              // log2(ow*ow)
    const int t = tok >> sh, r = tok & ((1 << sh) - 1);
    const int oy = r >> (5 - ls), ox = r & (ow - 1);
    const int cl = dd >> (2 * ls), pp = dd & ((1 << (2 * ls)) - 1);
    const int py = pp >> ls, px = pp & ((1 << ls) - 1);
    const int cg = (cfg << 9) + cl;
    const int y = (oy << ls) + py, x = (ox << ls) + px;
    const int bt = (b << 2) + t;
    ylo[((((size_t)bt << 10) + cg) << 10) + (y << 5) + x] = bv;
  } else {
    const int tk = tok - n_lo;
    const int lsh = ls + 1;                   // hi patch log: 3 / 2
    const int sh = 2 * (5 - ls);
    const int t = tk >> sh, r = tk & ((1 << sh) - 1);
    const int oy = r >> (5 - ls), ox = r & (ow - 1);
    const int cl = dd >> (2 * lsh), pp = dd & ((1 << (2 * lsh)) - 1);
    const int py = pp >> lsh, px = pp & ((1 << lsh) - 1);
    const int cg = (cfg << 7) + cl;
    const int y = (oy << lsh) + py, x = (ox << lsh) + px;
    const int bt = (b << 2) + t;
    yhi[((((size_t)bt << 8) + cg) << 12) + (y << 6) + x] = bv;
  }
}

// ---------------------------------------------------------------------------
// Kernels
// ---------------------------------------------------------------------------
__global__ void k_cvt(const float* __restrict__ x, bf16* __restrict__ y, int n) {
  int i = blockIdx.x * 256 + threadIdx.x;
  if (i < n) y[i] = (bf16)x[i];
}

// x [BT][C][HW] f32 -> xt [BT][HW][C] bf16 (LDS-tiled transpose)
__global__ void k_transpose_cvt(const float* __restrict__ X, bf16* __restrict__ Xt,
                                int C, int HW) {
  __shared__ float tile[32][33];
  const int bt = blockIdx.z;
  const int hw0 = blockIdx.x * 32, c0 = blockIdx.y * 32;
  const float* src = X + (size_t)bt * C * HW;
#pragma unroll
  for (int i = 0; i < 4; ++i) {
    int c = c0 + threadIdx.y + i * 8;
    tile[threadIdx.y + i * 8][threadIdx.x] = src[(size_t)c * HW + hw0 + threadIdx.x];
  }
  __syncthreads();
  bf16* dst = Xt + (size_t)bt * HW * C;
#pragma unroll
  for (int i = 0; i < 4; ++i) {
    int hw = hw0 + threadIdx.y + i * 8;
    dst[(size_t)hw * C + c0 + threadIdx.x] = (bf16)tile[threadIdx.x][threadIdx.y + i * 8];
  }
}

// QKV 1x1-conv GEMM + bias + fused to_patches scatter
__global__ void k_gemm_qkv(const bf16* __restrict__ A, int lda,
                           const bf16* __restrict__ B, int ldb, int K,
                           const float* __restrict__ bias, bf16* o0, bf16* o1,
                           int which, int stream) {
  v8f acc[2][2];
  zero_acc(acc);
  gemm_core(A, lda, B, ldb, K, acc);
  epilogue_apply(acc, [&](int m, int n, float v) {
    if (stream == 0) scatter_qkv_lo(m, n, v, bias, o0, o1, which);
    else             scatter_qkv_hi(m, n, v, bias, o0, o1, which);
  });
}

// S[b][qtok][ktok] = scale * Q[qtok]·K[ktok]
__global__ void k_gemm_scores(const bf16* __restrict__ Q, const bf16* __restrict__ Km,
                              int D, int T, float scale, float* __restrict__ S) {
  const int z = blockIdx.z;
  const bf16* Az = Q  + (size_t)z * T * D;
  const bf16* Bz = Km + (size_t)z * T * D;
  v8f acc[2][2];
  zero_acc(acc);
  gemm_core(Az, D, Bz, D, D, acc);
  epilogue_apply(acc, [&](int m, int n, float v) {
    S[((size_t)z * T + m) * T + n] = v * scale;
  });
}

// row-wise softmax, fp32 in -> bf16 out
__global__ void k_softmax(const float* __restrict__ S, bf16* __restrict__ P, int T) {
  const size_t row = blockIdx.x;
  const float* s = S + row * T;
  bf16* p = P + row * T;
  __shared__ float red[8];
  const int tid = threadIdx.x;
  float mx = -3.0e38f;
  for (int i = tid; i < T; i += 256) mx = fmaxf(mx, s[i]);
#pragma unroll
  for (int o = 16; o > 0; o >>= 1) mx = fmaxf(mx, __shfl_xor(mx, o, 32));
  if ((tid & 31) == 0) red[tid >> 5] = mx;
  __syncthreads();
  if (tid == 0) {
    float v = red[0];
#pragma unroll
    for (int i = 1; i < 8; ++i) v = fmaxf(v, red[i]);
    red[0] = v;
  }
  __syncthreads();
  mx = red[0];
  __syncthreads();
  float sum = 0.f;
  for (int i = tid; i < T; i += 256) sum += __expf(s[i] - mx);
#pragma unroll
  for (int o = 16; o > 0; o >>= 1) sum += __shfl_xor(sum, o, 32);
  if ((tid & 31) == 0) red[tid >> 5] = sum;
  __syncthreads();
  if (tid == 0) {
    float v = 0.f;
#pragma unroll
    for (int i = 0; i < 8; ++i) v += red[i];
    red[0] = v;
  }
  __syncthreads();
  const float inv = 1.0f / red[0];
  for (int i = tid; i < T; i += 256) p[i] = (bf16)(__expf(s[i] - mx) * inv);
}

// Y[tok][dd] = P[tok][:] x V ; V held transposed [dd][ktok]; fused from_patches
__global__ void k_gemm_pv(const bf16* __restrict__ Pm, const bf16* __restrict__ Vt,
                          int T, int D, bf16* ylo, bf16* yhi, int cfg) {
  const int z = blockIdx.z;
  const bf16* Az = Pm + (size_t)z * T * T;
  const bf16* Bz = Vt + (size_t)z * D * T;
  v8f acc[2][2];
  zero_acc(acc);
  gemm_core(Az, T, Bz, T, T, acc);
  epilogue_apply(acc, [&](int m, int n, float v) {
    scatter_pv(z, m, n, v, ylo, yhi, cfg);
  });
}

// 3x3 conv (pad 1) as implicit-im2col GEMM + bias + leaky-relu -> f32 NCHW out
__global__ void k_conv3x3(const bf16* __restrict__ X, const bf16* __restrict__ Wt,
                          const float* __restrict__ bias, float* __restrict__ out,
                          int Cin, int logW) {
  const int logHW = 2 * logW;
  const int W = 1 << logW, H = W, HW = 1 << logHW;
  const int K = Cin * 9;
  __shared__ __align__(16) bf16 ldsA[BM * LDT];
  __shared__ __align__(16) bf16 ldsB[BN * LDT];
  const int tid = threadIdx.x;
  const int lane = tid & 31, wid = tid >> 5;
  const int wm = wid & 3, wn = wid >> 2;
  const int m0 = blockIdx.y * BM, n0 = blockIdx.x * BN;
  const int br = tid >> 2, bs = (tid & 3) * 8;
  const int frow = lane & 15;
  const int fkbA = (lane >> 4) << 3, fkbB = (lane >> 4) << 3;
  const unsigned bDst = (unsigned)(uintptr_t)&ldsB[br * LDT + bs];
  v8f acc[2][2];
  zero_acc(acc);

  for (int k0 = 0; k0 < K; k0 += BK) {
    __syncthreads();
    // weight tile via async global->LDS copy (overlaps the im2col gather)
    async_b128(bDst, Wt + (size_t)(n0 + br) * K + k0 + bs);
#pragma unroll
    for (int e = 0; e < 16; ++e) {          // im2col gather: 128x32 tile
      int idx = e * 256 + tid;
      int row = idx >> 5, kk = idx & 31;
      int m = m0 + row;
      int bt = m >> logHW, hw = m & (HW - 1);
      int y = hw >> logW, x = hw & (W - 1);
      int k = k0 + kk;
      int ci = k / 9, r9 = k - ci * 9;
      int dy = r9 / 3, dx = r9 - dy * 3;
      int yy = y + dy - 1, xx = x + dx - 1;
      bf16 val = (bf16)0.0f;
      if ((unsigned)yy < (unsigned)H && (unsigned)xx < (unsigned)W)
        val = X[(((size_t)bt * Cin + ci) << logHW) + ((size_t)yy << logW) + xx];
      ldsA[row * LDT + kk] = val;
    }
    wait_async0();
    __syncthreads();

    v16bf a0 = frag16(&ldsA[(wm * 32 +      frow) * LDT + fkbA]);
    v16bf a1 = frag16(&ldsA[(wm * 32 + 16 + frow) * LDT + fkbA]);
    v16bf b0 = frag16(&ldsB[(wn * 32 +      frow) * LDT + fkbB]);
    v16bf b1 = frag16(&ldsB[(wn * 32 + 16 + frow) * LDT + fkbB]);
    acc[0][0] = wmma_bf16(a0, b0, acc[0][0]);
    acc[0][1] = wmma_bf16(a0, b1, acc[0][1]);
    acc[1][0] = wmma_bf16(a1, b0, acc[1][0]);
    acc[1][1] = wmma_bf16(a1, b1, acc[1][1]);
  }

  epilogue_apply(acc, [&](int m, int n, float v) {
    v += bias[n];
    v = v > 0.f ? v : 0.2f * v;
    int bt = m >> logHW, hw = m & (HW - 1);
    out[(((size_t)bt * 256 + n) << logHW) + hw] = v;
  });
}

// ---------------------------------------------------------------------------
// Host orchestration
// ---------------------------------------------------------------------------
extern "C" void kernel_launch(void* const* d_in, const int* in_sizes, int n_in,
                              void* d_out, int out_size, void* d_ws, size_t ws_size,
                              hipStream_t stream) {
  (void)in_sizes; (void)n_in; (void)out_size; (void)ws_size;
  const float* x_lo  = (const float*)d_in[0];
  const float* x_hi  = (const float*)d_in[1];
  const float* wq_lo = (const float*)d_in[3];  const float* bq_lo = (const float*)d_in[4];
  const float* wk_lo = (const float*)d_in[5];  const float* bk_lo = (const float*)d_in[6];
  const float* wv_lo = (const float*)d_in[7];  const float* bv_lo = (const float*)d_in[8];
  const float* wq_hi = (const float*)d_in[9];  const float* bq_hi = (const float*)d_in[10];
  const float* wk_hi = (const float*)d_in[11]; const float* bk_hi = (const float*)d_in[12];
  const float* wv_hi = (const float*)d_in[13]; const float* bv_hi = (const float*)d_in[14];
  const float* wo_lo = (const float*)d_in[15]; const float* bo_lo = (const float*)d_in[16];
  const float* wo_hi = (const float*)d_in[17]; const float* bo_hi = (const float*)d_in[18];
  float* out = (float*)d_out;

  char* ws = (char*)d_ws;
  size_t off = 0;
  auto alloc = [&](size_t bytes) -> void* {
    void* p = ws + off;
    off = (off + bytes + 255) & ~(size_t)255;
    return p;
  };

  const size_t E = 16777216;  // 4*2048*2048 = 4*512*8192
  bf16* XloT = (bf16*)alloc((size_t)4194304 * 2);
  bf16* XhiT = (bf16*)alloc(E * 2);
  bf16* WqLo = (bf16*)alloc(262144 * 2);
  bf16* WkLo = (bf16*)alloc(262144 * 2);
  bf16* WvLo = (bf16*)alloc(262144 * 2);
  bf16* WqHi = (bf16*)alloc(65536 * 2);
  bf16* WkHi = (bf16*)alloc(65536 * 2);
  bf16* WvHi = (bf16*)alloc(65536 * 2);
  bf16* WoLo = (bf16*)alloc((size_t)2359296 * 2);
  bf16* WoHi = (bf16*)alloc((size_t)589824 * 2);
  bf16* Q0  = (bf16*)alloc(E * 2); bf16* Q1  = (bf16*)alloc(E * 2);
  bf16* K0m = (bf16*)alloc(E * 2); bf16* K1m = (bf16*)alloc(E * 2);
  bf16* V0t = (bf16*)alloc(E * 2); bf16* V1t = (bf16*)alloc(E * 2);
  float* S  = (float*)alloc(E * 4);
  bf16* P   = (bf16*)alloc(E * 2);
  bf16* YloG = (bf16*)alloc(E * 2);
  bf16* YhiG = (bf16*)alloc(E * 2);

  auto cvt = [&](const float* s, bf16* d, int n) {
    k_cvt<<<dim3((n + 255) / 256), dim3(256), 0, stream>>>(s, d, n);
  };
  cvt(wq_lo, WqLo, 262144); cvt(wk_lo, WkLo, 262144); cvt(wv_lo, WvLo, 262144);
  cvt(wq_hi, WqHi, 65536);  cvt(wk_hi, WkHi, 65536);  cvt(wv_hi, WvHi, 65536);
  cvt(wo_lo, WoLo, 2359296); cvt(wo_hi, WoHi, 589824);

  k_transpose_cvt<<<dim3(32, 8, 16), dim3(32, 8), 0, stream>>>(x_lo, XloT, 256, 1024);
  k_transpose_cvt<<<dim3(128, 8, 16), dim3(32, 8), 0, stream>>>(x_hi, XhiT, 256, 4096);

  // QKV projections (1x1 conv) with fused to_patches scatter
  dim3 gLo(16, 128), gHi(4, 512);
  k_gemm_qkv<<<gLo, 256, 0, stream>>>(XloT, 256, WqLo, 256, 256, bq_lo, Q0, Q1, 0, 0);
  k_gemm_qkv<<<gLo, 256, 0, stream>>>(XloT, 256, WkLo, 256, 256, bk_lo, K0m, K1m, 1, 0);
  k_gemm_qkv<<<gLo, 256, 0, stream>>>(XloT, 256, WvLo, 256, 256, bv_lo, V0t, V1t, 2, 0);
  k_gemm_qkv<<<gHi, 256, 0, stream>>>(XhiT, 256, WqHi, 256, 256, bq_hi, Q0, Q1, 0, 1);
  k_gemm_qkv<<<gHi, 256, 0, stream>>>(XhiT, 256, WkHi, 256, 256, bk_hi, K0m, K1m, 1, 1);
  k_gemm_qkv<<<gHi, 256, 0, stream>>>(XhiT, 256, WvHi, 256, 256, bv_hi, V0t, V1t, 2, 1);

  // attention, config 0 (patch 4x4): T=512, D=8192
  const float sc0 = 1.0f / sqrtf(8192.0f);
  k_gemm_scores<<<dim3(8, 4, 4), 256, 0, stream>>>(Q0, K0m, 8192, 512, sc0, S);
  k_softmax<<<dim3(2048), 256, 0, stream>>>(S, P, 512);
  k_gemm_pv<<<dim3(128, 4, 4), 256, 0, stream>>>(P, V0t, 512, 8192, YloG, YhiG, 0);

  // attention, config 1 (patch 2x2): T=2048, D=2048
  const float sc1 = 1.0f / sqrtf(2048.0f);
  k_gemm_scores<<<dim3(32, 16, 4), 256, 0, stream>>>(Q1, K1m, 2048, 2048, sc1, S);
  k_softmax<<<dim3(8192), 256, 0, stream>>>(S, P, 2048);
  k_gemm_pv<<<dim3(32, 16, 4), 256, 0, stream>>>(P, V1t, 2048, 2048, YloG, YhiG, 1);

  // output 3x3 convs + leaky relu -> d_out (out_lo then out_hi)
  k_conv3x3<<<dim3(4, 128), 256, 0, stream>>>(YloG, WoLo, bo_lo, out, 1024, 5);
  k_conv3x3<<<dim3(4, 512), 256, 0, stream>>>(YhiG, WoHi, bo_hi, out + 4194304, 256, 6);
}
